// Block_44238163149136
// MI455X (gfx1250) — compile-verified, matrix operand order checked by reading, not run on hardware
//
#include <hip/hip_runtime.h>
#include <hip/hip_bf16.h>

typedef __attribute__((ext_vector_type(16))) _Float16 v16h;
typedef __attribute__((ext_vector_type(8)))  _Float16 v8h;
typedef __attribute__((ext_vector_type(4)))  _Float16 v4h;
typedef __attribute__((ext_vector_type(8)))  float    v8f;

// Problem constants
#define BB   8
#define LL   4096
#define DIMM 256
#define DII  512
#define KK   2
#define NN   16
#define RR   16
#define MTOK (BB * LL)   // 32768 tokens

__device__ __forceinline__ v4h pack4(float4 v) {
    return (v4h){(_Float16)v.x, (_Float16)v.y, (_Float16)v.z, (_Float16)v.w};
}

// ---------------------------------------------------------------------------
// Unified token-major WMMA GEMM:  C[m, n] = sum_k A[m, k] * W[n, k]  (+epilogue)
// Block: 256 threads = 8 waves (2 m-tiles x 4 n-tiles) -> 32x64 output tile.
// Staging: each thread loads one float4 (b128) per tile slice, converts to
// f16, ds_store_b64 into LDS (stride 40 halves = 80B: 16B-aligned, bank-
// spread). Fragment gather: 2x ds_load_b128 per matrix per the CDNA5 16-bit
// layout (lane's halves live in K-groups [8*hi,8*hi+8) and [16+8*hi,+8)).
//
// MODE 0: in_proj   A=x[32768,256]      W=W_in[1024,256]  -> xx_raw | silu(z)
// MODE 1: conv      A=xx_raw[.,512]     W=conv_w[512,512] -> silu(v + conv_b)
// MODE 2: x_proj    A=xconv (flip if k) W=x_proj_w[k]     -> xdbl[b,k,l,48]
// MODE 3: dt_proj   A=xdbl[b,k,l,48]    W=dt_w[k][512,16] -> softplus(v+dt_b)
// MODE 4: out_proj  A=yln[.,512]        W=W_out[256,512]  -> d_out
// ---------------------------------------------------------------------------
template <int MODE>
__global__ __launch_bounds__(256) void gemm_wmma(
    const float* __restrict__ A, const float* __restrict__ W,
    const float* __restrict__ bias, float* __restrict__ C0,
    float* __restrict__ C1, int M, int N, int Kdim, int lda, int ldw, int L)
{
    constexpr int BM = 32, BN = 64, KC = 32, STR = 40; // STR halves per row
    __shared__ _Float16 sAh[BM * STR];  // 2560 B
    __shared__ _Float16 sWh[BN * STR];  // 5120 B

    const int tiles_n = (N + BN - 1) / BN;
    const int bm = blockIdx.x / tiles_n;
    const int bn = blockIdx.x % tiles_n;
    const int kg = blockIdx.y;          // direction group (modes 2,3)
    const int m0 = bm * BM;
    const int n0 = bn * BN;
    const int tid  = threadIdx.x;
    const int lane = tid & 31;
    const int wave = tid >> 5;
    const int wm = wave >> 2, wn = wave & 3;
    const bool lo = lane < 16;
    const int  mr  = lane & 15;
    const int  hi8 = lo ? 0 : 8;

    const float* Wp = W;
    const float* bp = bias;
    if constexpr (MODE == 2) { Wp = W + (long)kg * 48 * DII; }
    if constexpr (MODE == 3) { Wp = W + (long)kg * DII * RR; bp = bias + kg * DII; }

    // ---- per-thread staging coordinates (uniform across K loop) ----
    const int rowA = tid >> 3;           // 0..31
    const int colA = (tid & 7) * 4;      // 0,4,...,28
    long arow;
    {
        const int m = m0 + rowA;
        if constexpr (MODE == 2) {
            const int b = m / L, l = m % L;
            arow = (long)b * L + (kg ? (L - 1 - l) : l);
        } else if constexpr (MODE == 3) {
            const int b = m / L, l = m % L;
            arow = ((long)b * KK + kg) * L + l;
        } else {
            arow = m;
        }
    }
    const float* aptr = A + arow * lda + colA;

    const int rowW0 = tid >> 3;          // 0..31
    const int rowW1 = rowW0 + 32;        // 32..63
    const int colW  = (tid & 7) * 4;
    const float* wptr0 = Wp + (long)(n0 + rowW0) * ldw + colW;
    const float* wptr1 = Wp + (long)(n0 + rowW1) * ldw + colW;
    const bool w1_ok = (MODE != 2) || (n0 + rowW1 < N);   // N=48 tail rows
    const bool k_ok  = (MODE != 3) || (colW < RR);        // K=16 zero-pad

    v8f acc = {0.f, 0.f, 0.f, 0.f, 0.f, 0.f, 0.f, 0.f};

    for (int k0 = 0; k0 < Kdim; k0 += KC) {
        // ---- stage A (32x32 f32 -> f16), one b128 load + one b64 LDS store --
        {
            float4 va = k_ok ? *(const float4*)(aptr + k0)
                             : float4{0.f, 0.f, 0.f, 0.f};
            *(v4h*)(sAh + rowA * STR + colA) = pack4(va);
        }
        // ---- stage W (64x32 f32 -> f16), two b128 loads ----
        {
            float4 v0 = k_ok ? *(const float4*)(wptr0 + k0)
                             : float4{0.f, 0.f, 0.f, 0.f};
            float4 v1 = (k_ok && w1_ok) ? *(const float4*)(wptr1 + k0)
                                        : float4{0.f, 0.f, 0.f, 0.f};
            *(v4h*)(sWh + rowW0 * STR + colW) = pack4(v0);
            *(v4h*)(sWh + rowW1 * STR + colW) = pack4(v1);
        }
        // prefetch next K-slice while this one is consumed
        if (k0 + KC < Kdim) {
            __builtin_prefetch(aptr + k0 + KC, 0, 0);
            __builtin_prefetch(wptr0 + k0 + KC, 0, 0);
        }
        __syncthreads();

        // ---- fragment gather: 2x ds_load_b128 per matrix ----
        const int rA = wm * 16 + mr;
        const int rB = wn * 16 + mr;
        v8h aL = *(const v8h*)(sAh + rA * STR + hi8);
        v8h aH = *(const v8h*)(sAh + rA * STR + 16 + hi8);
        v8h bL = *(const v8h*)(sWh + rB * STR + hi8);
        v8h bH = *(const v8h*)(sWh + rB * STR + 16 + hi8);
        v16h af = __builtin_shufflevector(aL, aH, 0, 1, 2, 3, 4, 5, 6, 7,
                                          8, 9, 10, 11, 12, 13, 14, 15);
        v16h bf = __builtin_shufflevector(bL, bH, 0, 1, 2, 3, 4, 5, 6, 7,
                                          8, 9, 10, 11, 12, 13, 14, 15);
        acc = __builtin_amdgcn_wmma_f32_16x16x32_f16(
            false, af, false, bf, (short)0, acc, false, false);
        __syncthreads();
    }

    // ---- epilogue: C/D layout -> VGPR r holds row (lo ? r : r+8), col = mr --
#pragma unroll
    for (int r = 0; r < 8; ++r) {
        const int m = m0 + wm * 16 + (lo ? r : r + 8);
        const int n = n0 + wn * 16 + mr;
        if (MODE == 2 && n >= N) continue;
        float v = acc[r];
        if constexpr (MODE == 0) {
            if (n < DII) {
                C0[(long)m * DII + n] = v;                       // xx (pre-conv)
            } else {
                const float s = v / (1.f + __expf(-v));          // silu(z)
                C1[(long)m * DII + (n - DII)] = s;
            }
        } else if constexpr (MODE == 1) {
            v += bias[n];
            v = v / (1.f + __expf(-v));                          // silu
            C0[(long)m * DII + n] = v;
        } else if constexpr (MODE == 2) {
            const int b = m / L, l = m % L;
            C0[(((long)b * KK + kg) * L + l) * 48 + n] = v;
        } else if constexpr (MODE == 3) {
            const int b = m / L, l = m % L;
            v += bp[n];
            const float sp = (v > 20.f) ? v : log1pf(__expf(v)); // softplus
            C0[(((long)b * KK + kg) * L + l) * DII + n] = sp;
        } else { // MODE 4
            C0[(long)m * DIMM + n] = v;
        }
    }
}

// ---------------------------------------------------------------------------
// Selective scan: one thread per (b, k, d) channel, N=16 states in registers.
// Block = 256 threads, all sharing one (b,k). Bs/Cs (32 f32 per step) are
// broadcast via LDS; staging uses the gfx1250 async global->LDS engine
// (global_load_async_to_lds_b32, ASYNCcnt) with double buffering: issue the
// next chunk, then s_wait_asynccnt 1 so only the *older* (current) chunk is
// required complete -- async loads retire in order, hiding the next chunk's
// HBM latency behind the 8 serial scan steps.
// ---------------------------------------------------------------------------
__global__ __launch_bounds__(256) void scan_kernel(
    const float* __restrict__ xconv, const float* __restrict__ xdbl,
    const float* __restrict__ delta, const float* __restrict__ A_logs,
    const float* __restrict__ Ds, float* __restrict__ ydir, int L)
{
    constexpr int CH = 8;
    __shared__ float sbc[2][CH][32];   // [buf][step][0..15]=Bs, [16..31]=Cs

    const int tid   = threadIdx.x;
    const int chunk = blockIdx.x & 1;
    const int k     = (blockIdx.x >> 1) & 1;
    const int b     = blockIdx.x >> 2;
    const int d     = chunk * 256 + tid;

    float a[NN], h[NN];
#pragma unroll
    for (int n = 0; n < NN; ++n) {
        a[n] = -__expf(A_logs[((long)k * DII + d) * NN + n]);
        h[n] = 0.f;
    }
    const float dD = Ds[k * DII + d];
    const long base_tok = ((long)b * KK + k) * L;

    const int ls = tid >> 5, lc = tid & 31;   // this thread's (step, col) slot
    auto issue_chunk = [&](int buf, int t0) {
        const unsigned long long g =
            (unsigned long long)(uintptr_t)&xdbl[(base_tok + t0 + ls) * 48 + RR + lc];
        const unsigned lds = (unsigned)(uintptr_t)&sbc[buf][ls][lc];
        asm volatile("global_load_async_to_lds_b32 %0, %1, off"
                     :: "v"(lds), "v"(g) : "memory");
    };

    issue_chunk(0, 0);                 // prime buffer 0
    int buf = 0;
    for (int t0 = 0; t0 < L; t0 += CH) {
        if (t0 + CH < L) {
            issue_chunk(buf ^ 1, t0 + CH);                 // 2 outstanding
            asm volatile("s_wait_asynccnt 1" ::: "memory"); // older one done
        } else {
            asm volatile("s_wait_asynccnt 0" ::: "memory");
        }
        __syncthreads();
        for (int s = 0; s < CH; ++s) {
            const int  t   = t0 + s;
            const long tok = base_tok + t;
            const float dt = delta[tok * DII + d];
            const int  lu  = k ? (L - 1 - t) : t;
            const float u  = xconv[((long)b * L + lu) * DII + d];
            const float du = dt * u;
            float accy = 0.f;
#pragma unroll
            for (int n = 0; n < NN; ++n) {
                const float e = __expf(dt * a[n]);
                h[n] = e * h[n] + du * sbc[buf][s][n];
                accy += h[n] * sbc[buf][s][16 + n];
            }
            ydir[tok * DII + d] = accy + dD * u;
        }
        __syncthreads();               // all reads done before buf is re-filled
        buf ^= 1;
    }
}

// ---------------------------------------------------------------------------
// Direction merge (k=1 read at flipped l) + LayerNorm(DI) + z-gate.
// One block per token; each thread owns 2 of the 512 channels.
// ---------------------------------------------------------------------------
__global__ __launch_bounds__(256) void ln_mul_kernel(
    const float* __restrict__ ydir, const float* __restrict__ z,
    const float* __restrict__ ln_w, const float* __restrict__ ln_b,
    float* __restrict__ yln, int L)
{
    __shared__ float ssum[256], ssq[256];
    const int token = blockIdx.x;
    const int b = token / L, l = token % L;
    const int tid = threadIdx.x;

    const long t0 = ((long)b * KK + 0) * L + l;            // forward
    const long t1 = ((long)b * KK + 1) * L + (L - 1 - l);  // backward, flipped
    const float v0 = ydir[t0 * DII + tid]       + ydir[t1 * DII + tid];
    const float v1 = ydir[t0 * DII + tid + 256] + ydir[t1 * DII + tid + 256];

    ssum[tid] = v0 + v1;
    ssq[tid]  = v0 * v0 + v1 * v1;
    __syncthreads();
    for (int off = 128; off > 0; off >>= 1) {
        if (tid < off) { ssum[tid] += ssum[tid + off]; ssq[tid] += ssq[tid + off]; }
        __syncthreads();
    }
    const float mu   = ssum[0] * (1.f / DII);
    const float var  = ssq[0] * (1.f / DII) - mu * mu;
    const float rstd = rsqrtf(var + 1e-5f);

    const long zt = (long)token * DII;
    yln[zt + tid]       = ((v0 - mu) * rstd * ln_w[tid]       + ln_b[tid])       * z[zt + tid];
    yln[zt + tid + 256] = ((v1 - mu) * rstd * ln_w[tid + 256] + ln_b[tid + 256]) * z[zt + tid + 256];
}

// ---------------------------------------------------------------------------
extern "C" void kernel_launch(void* const* d_in, const int* in_sizes, int n_in,
                              void* d_out, int out_size, void* d_ws, size_t ws_size,
                              hipStream_t stream)
{
    const float* x       = (const float*)d_in[0];
    const float* W_in    = (const float*)d_in[1];
    const float* conv_w  = (const float*)d_in[2];
    const float* conv_b  = (const float*)d_in[3];
    const float* x_projw = (const float*)d_in[4];
    const float* dt_w    = (const float*)d_in[5];
    const float* dt_b    = (const float*)d_in[6];
    const float* A_logs  = (const float*)d_in[7];
    const float* Ds      = (const float*)d_in[8];
    const float* ln_w    = (const float*)d_in[9];
    const float* ln_b    = (const float*)d_in[10];
    const float* W_out   = (const float*)d_in[11];
    float* out = (float*)d_out;

    // Workspace carve-up (all f32, all fully written before read)
    char* ws = (char*)d_ws;
    size_t off = 0;
    auto carve = [&](size_t elems) {
        float* p = (float*)(ws + off);
        off += ((elems * sizeof(float)) + 255) & ~size_t(255);
        return p;
    };
    float* xx_raw = carve((size_t)MTOK * DII);            // pre-conv xx
    float* zbuf   = carve((size_t)MTOK * DII);            // silu(z)
    float* xconv  = carve((size_t)MTOK * DII);            // silu(conv(xx))
    float* xdbl   = carve((size_t)BB * KK * LL * 48);     // [b,k,l,48]
    float* deltab = carve((size_t)BB * KK * LL * DII);    // softplus(dt)
    float* ydir   = carve((size_t)BB * KK * LL * DII);    // per-direction y
    float* yln    = carve((size_t)MTOK * DII);            // LN(y)*z

    const dim3 blk(256);
    const int MT = MTOK;

    // 1) in_proj: [32768,256] x [1024,256]^T -> xx_raw | silu(z)
    gemm_wmma<0><<<dim3((MT / 32) * (1024 / 64), 1), blk, 0, stream>>>(
        x, W_in, nullptr, xx_raw, zbuf, MT, 1024, DIMM, DIMM, DIMM, LL);

    // 2) conv (1x1): [32768,512] x [512,512]^T -> silu(+conv_b)
    gemm_wmma<1><<<dim3((MT / 32) * (DII / 64), 1), blk, 0, stream>>>(
        xx_raw, conv_w, conv_b, xconv, nullptr, MT, DII, DII, DII, DII, LL);

    // 3) x_proj per direction (k=1 reads flipped tokens): -> xdbl[b,k,l,48]
    gemm_wmma<2><<<dim3((MT / 32) * 1, KK), blk, 0, stream>>>(
        xconv, x_projw, nullptr, xdbl, nullptr, MT, 48, DII, DII, DII, LL);

    // 4) dt_proj (K=16 zero-padded to 32) + softplus(+dt_b): -> delta
    gemm_wmma<3><<<dim3((MT / 32) * (DII / 64), KK), blk, 0, stream>>>(
        xdbl, dt_w, dt_b, deltab, nullptr, MT, DII, RR, 48, RR, LL);

    // 5) selective scan: 32 blocks x 256 threads = 8192 (b,k,d) channels
    scan_kernel<<<dim3(BB * KK * 2), blk, 0, stream>>>(
        xconv, xdbl, deltab, A_logs, Ds, ydir, LL);

    // 6) direction merge + LayerNorm + z gate
    ln_mul_kernel<<<dim3(MT), blk, 0, stream>>>(ydir, zbuf, ln_w, ln_b, yln, LL);

    // 7) out_proj: [32768,512] x [256,512]^T -> d_out
    gemm_wmma<4><<<dim3((MT / 32) * (DIMM / 64), 1), blk, 0, stream>>>(
        yln, W_out, nullptr, out, nullptr, MT, DIMM, DII, DII, DII, LL);
}
